// SurfaceBind_56453050138692
// MI455X (gfx1250) — matrix operand order repaired and checked by reading, not compiled, not run on hardware
//
#include <hip/hip_runtime.h>

// ============================================================================
// SurfaceBind pipeline for MI455X (gfx1250, wave32).
//
// All dense linear layers -> bf16-input WMMA GEMM (v_wmma_f32_16x16x32_bf16),
// f32 accumulate. Weights pre-packed (zero-padded to 4-column-tile groups) to
// the per-lane B fragment layout so the B operand is two contiguous uint4
// loads per lane per K-step and the inner loop is completely branch-free:
// each wave runs 4 unconditional WMMAs per K-step (16x64 slab) off one shared
// A fragment staged through LDS by the whole block. Graph scatter/segment
// sums use relaxed agent-scope f32 atomics (served by the 192MB L2).
//
// Input flattening assumption (jax tree flatten: dict keys sorted at every
// level, lists in order):
//  0 atom_cat  1 atom_dist  2 edge_attr  3 edge_index  4 neg_pep_idx
//  5 neg_prot_idx  6..158 params leaves  159 peptide_edge_index
//  160 peptide_pretrained  161 pos  162 pos_pep_idx  163 pos_prot_idx 164 x_geom
// params leaves (alphabetical): aanet_proj(6..13) atom_net_a(14..21)
//  atom_net_b(22..29) chemical_net(30..37) edge_nets(38..57) emb(58)
//  feat_scale(59,60) gins(61..130; per gin: lin1a.W,b lin1b.W,b lin2a.W,b
//  lin2b.W,b ln1.b,g ln2.b,g ln_mid.b,g) in_net(131,132) out_net(133,134)
//  pep_gin(135..140: lin_a.W,b lin_b.W,b ln.b,g) pro_linear(141..150)
//  surface_mask_net(151..158)
// d_out: pos_pred[50000] | neg_pred[50000] | mp_pos[50000*80] | mp_neg[50000*80]
// ============================================================================

#define NATOMS 100000
#define KNBR   16
#define NEDGES 1600000
#define MPEP   20000
#define EPEP   80000
#define NPAIRS 50000

#define ACT_NONE 0
#define ACT_RELU 1
#define ACT_SIG  2

typedef __attribute__((ext_vector_type(16))) __bf16 v16bf;
typedef __attribute__((ext_vector_type(8)))  float  v8f;

static __device__ __forceinline__ unsigned short f32_to_bf16(float f) {
  union { float f; unsigned int u; } cv; cv.f = f;
  return (unsigned short)((cv.u + 0x7FFFu + ((cv.u >> 16) & 1u)) >> 16);
}

static __device__ __forceinline__ void atomic_add_f32(float* p, float v) {
  (void)__hip_atomic_fetch_add(p, v, __ATOMIC_RELAXED, __HIP_MEMORY_SCOPE_AGENT);
}

// ---------------------------------------------------------------------------
// Pack f32 row-major W[K x N] into bf16 WMMA B fragments; column tiles padded
// to a multiple of 4 (zero fill) so the GEMM inner loop needs no tile guards.
// Fragment layout (wave32, 16x16x32): lane l serves column (l&15); its 16
// bf16 elements cover K = kt*32 + (l>>4)*16 .. +15 in order, contiguous in
// memory -> two aligned uint4 loads per lane in the GEMM inner loop.
// ---------------------------------------------------------------------------
__global__ void pack_w_kernel(const float* __restrict__ W, unsigned short* __restrict__ P,
                              int K, int N, int nKT, long total) {
  long t = (long)blockIdx.x * 256 + threadIdx.x;
  if (t >= total) return;
  int  i   = (int)(t & 15);
  int  l   = (int)((t >> 4) & 31);
  long blk = t >> 9;
  int  kt  = (int)(blk % nKT);
  int  ct  = (int)(blk / nKT);
  int k = kt * 32 + ((l >> 4) << 4) + i;
  int c = ct * 16 + (l & 15);
  float v = (k < K && c < N) ? W[(size_t)k * N + c] : 0.0f;
  P[t] = f32_to_bf16(v);
}

// ---------------------------------------------------------------------------
// D[MxN] = act(A[MxK] (+A2) @ W + bias), bf16 WMMA, f32 accumulate.
// Block: 16-row A tile staged in LDS; each of the 4 waves owns a group of 4
// consecutive 16-col tiles (16x64 slab): 4 unconditional WMMAs per K-step per
// wave from one shared A fragment (pack buffer is tile-padded). A-fragment
// per lane: K = (lane>>4)*16 .. +15 contiguous, same order as the packed B.
// ---------------------------------------------------------------------------
template <int ACT, bool HASA2>
__global__ __launch_bounds__(128)
void gemm_bf16_wmma_kernel(const float* __restrict__ A, const float* __restrict__ A2,
                           const unsigned short* __restrict__ Wp,
                           const float* __restrict__ bias, float* __restrict__ D,
                           int M, int K, int N) {
  __shared__ float As[16 * 32];
  const int tid  = threadIdx.x;
  const int lane = tid & 31;
  const int wave = tid >> 5;
  const int m0   = blockIdx.x << 4;
  const int nNT  = (N + 15) >> 4;
  const int nKT  = (K + 31) >> 5;
  const int nGrp = (nNT + 3) >> 2;
  int grp = blockIdx.y * 4 + wave;
  if (grp >= nGrp) grp = nGrp - 1;   // uniform clamp; duplicate stores benign
  const int nt0 = grp * 4;           // first of 4 col tiles (all valid, padded)
  const int kh  = (lane >> 4) << 3;  // 0 or 8
  const size_t tileStride = (size_t)nKT * 512;
  const unsigned short* bbase = Wp + (size_t)nt0 * tileStride + (size_t)lane * 16;

  const v8f zero = {0.f, 0.f, 0.f, 0.f, 0.f, 0.f, 0.f, 0.f};
  v8f acc[4];
#pragma unroll
  for (int j = 0; j < 4; ++j) acc[j] = zero;

  for (int kt = 0; kt < nKT; ++kt) {
    __syncthreads();
#pragma unroll
    for (int u = 0; u < 4; ++u) {
      int idx = tid * 4 + u;                 // 0..511 covers the 16x32 A tile
      int r = idx >> 5, c = idx & 31;
      int m = m0 + r, k = kt * 32 + c;
      float v = 0.0f;
      if (m < M && k < K) {
        v = A[(size_t)m * K + k];
        if (HASA2) v += A2[(size_t)m * K + k];
        if (u == 0 && k + 32 < K) __builtin_prefetch(&A[(size_t)m * K + k + 32], 0, 1);
      }
      As[idx] = v;
    }
    __syncthreads();

    union { unsigned short us[16]; v16bf v; } af;
    const float* l0 = &As[(lane & 15) * 32 + ((lane >> 4) << 4)];
#pragma unroll
    for (int i = 0; i < 8; ++i) {
      af.us[i]     = f32_to_bf16(l0[i]);
      af.us[8 + i] = f32_to_bf16(l0[8 + i]);
    }

    const unsigned short* bk = bbase + (size_t)kt * 512;
#pragma unroll
    for (int j = 0; j < 4; ++j) {
      union { unsigned short us[16]; uint4 q[2]; v16bf v; } bf;
      const uint4* bq = (const uint4*)(bk + (size_t)j * tileStride);
      bf.q[0] = bq[0];
      bf.q[1] = bq[1];
      acc[j] = __builtin_amdgcn_wmma_f32_16x16x32_bf16(false, af.v, false, bf.v,
                                                       (short)0, acc[j], false, false);
    }
  }

#pragma unroll
  for (int j = 0; j < 4; ++j) {
    int c = ((nt0 + j) << 4) + (lane & 15);
    if (c < N) {                             // masks padded/ghost tiles too
      float bv = bias ? bias[c] : 0.0f;
#pragma unroll
      for (int v = 0; v < 8; ++v) {
        int m = m0 + v + kh;                 // C/D layout: lanes>=16 -> rows +8
        if (m < M) {
          float x = acc[j][v] + bv;
          if (ACT == ACT_RELU) x = fmaxf(x, 0.0f);
          if (ACT == ACT_SIG)  x = __builtin_amdgcn_rcpf(1.0f + __expf(-x));
          D[(size_t)m * N + c] = x;
        }
      }
    }
  }
}

// ---------------------------------------------------------------------------
// LayerNorm over last dim, one wave per row, optional fused ReLU.
// ---------------------------------------------------------------------------
__global__ __launch_bounds__(128)
void layernorm_kernel(const float* __restrict__ X, const float* __restrict__ gamma,
                      const float* __restrict__ beta, float* __restrict__ Y,
                      int M, int N, int relu) {
  int row  = blockIdx.x * 4 + (threadIdx.x >> 5);
  int lane = threadIdx.x & 31;
  if (row >= M) return;
  const float* x = X + (size_t)row * N;
  float s = 0.f, s2 = 0.f;
  for (int i = lane; i < N; i += 32) { float v = x[i]; s += v; s2 += v * v; }
  for (int o = 16; o > 0; o >>= 1) { s += __shfl_down(s, o); s2 += __shfl_down(s2, o); }
  s = __shfl(s, 0); s2 = __shfl(s2, 0);
  float mean = s / (float)N;
  float var  = s2 / (float)N - mean * mean;
  float inv  = rsqrtf(var + 1e-5f);
  float* y = Y + (size_t)row * N;
  for (int i = lane; i < N; i += 32) {
    float v = (x[i] - mean) * inv * gamma[i] + beta[i];
    if (relu) v = fmaxf(v, 0.f);
    y[i] = v;
  }
}

__global__ void fill_kernel(float* __restrict__ p, float v, long n) {
  long i = (long)blockIdx.x * 256 + threadIdx.x;
  if (i < n) p[i] = v;
}

// Agg[dst[e]] += X[src[e]]  (segment_sum of gathered rows)
__global__ void seg_gather_scatter_kernel(const float* __restrict__ X, const int* __restrict__ src,
                                          const int* __restrict__ dst, long E, int Dm,
                                          float* __restrict__ Agg) {
  long t = (long)blockIdx.x * 256 + threadIdx.x;
  if (t >= E * (long)Dm) return;
  long e = t / Dm; int d = (int)(t % Dm);
  atomic_add_f32(&Agg[(size_t)dst[e] * Dm + d], X[(size_t)src[e] * Dm + d]);
}

// Agg[idx[e]] += S[e]
__global__ void scatter_rows_kernel(const float* __restrict__ S, const int* __restrict__ idx,
                                    long n, int Dm, float* __restrict__ Agg) {
  long t = (long)blockIdx.x * 256 + threadIdx.x;
  if (t >= n * (long)Dm) return;
  long e = t / Dm; int d = (int)(t % Dm);
  atomic_add_f32(&Agg[(size_t)idx[e] * Dm + d], S[t]);
}

// X[(r,d)] for atom-neighbor rows: [emb[cat[r]] (32), 1/dist[r]]
__global__ void build_atom_input_kernel(const int* __restrict__ cat, const float* __restrict__ dist,
                                        const float* __restrict__ emb, float* __restrict__ X,
                                        int r0, int len) {
  long t = (long)blockIdx.x * 256 + threadIdx.x;
  if (t >= (long)len * 33) return;
  int r = (int)(t / 33), d = (int)(t % 33);
  int g = r0 + r;
  X[t] = (d < 32) ? emb[(size_t)cat[g] * 32 + d] : (1.0f / dist[g]);
}

// sum over the 16 neighbor rows of each atom (chunk aligned to 16)
__global__ void reduce16_kernel(const float* __restrict__ Y, float* __restrict__ out,
                                int r0, int len) {
  long t = (long)blockIdx.x * 256 + threadIdx.x;
  long total = (long)(len / 16) * 15;
  if (t >= total) return;
  int a = (int)(t / 15), d = (int)(t % 15);
  float s = 0.f;
#pragma unroll
  for (int j = 0; j < 16; ++j) s += Y[(size_t)(a * 16 + j) * 15 + d];
  out[(size_t)(r0 / 16 + a) * 15 + d] = s;
}

__global__ void build_feat16_kernel(const float* __restrict__ a2, const float* __restrict__ xg,
                                    float* __restrict__ o, int n) {
  long t = (long)blockIdx.x * 256 + threadIdx.x;
  if (t >= (long)n * 16) return;
  int r = (int)(t / 16), d = (int)(t % 16);
  o[t] = (d < 15) ? a2[(size_t)r * 15 + d] : xg[r];
}

__global__ void edge_dist_kernel(const float* __restrict__ pos, const int* __restrict__ src,
                                 const int* __restrict__ dst, float* __restrict__ out, int E) {
  int e = blockIdx.x * 256 + threadIdx.x;
  if (e >= E) return;
  int s = src[e], d = dst[e];
  float dx = pos[(size_t)s * 3 + 0] - pos[(size_t)d * 3 + 0];
  float dy = pos[(size_t)s * 3 + 1] - pos[(size_t)d * 3 + 1];
  float dz = pos[(size_t)s * 3 + 2] - pos[(size_t)d * 3 + 2];
  out[e] = sqrtf(dx * dx + dy * dy + dz * dz);
}

// Xe[e] = [dist, edge_attr, feat[dst[e]](80)]
__global__ void build_edge_input_kernel(const float* __restrict__ dist, const float* __restrict__ ea,
                                        const int* __restrict__ dst, const float* __restrict__ feat,
                                        float* __restrict__ X, int e0, int len) {
  long t = (long)blockIdx.x * 256 + threadIdx.x;
  if (t >= (long)len * 82) return;
  int e = (int)(t / 82), d = (int)(t % 82);
  int ge = e0 + e;
  float v;
  if (d == 0)      v = dist[ge];
  else if (d == 1) v = ea[ge];
  else             v = feat[(size_t)dst[ge] * 80 + (d - 2)];
  X[t] = v;
}

__global__ void ew_add_kernel(const float* __restrict__ a, const float* __restrict__ b,
                              float* __restrict__ y, long n) {
  long i = (long)blockIdx.x * 256 + threadIdx.x;
  if (i < n) y[i] = a[i] + b[i];
}

__global__ void ew_mul_kernel(const float* __restrict__ a, const float* __restrict__ b,
                              float* __restrict__ y, long n) {
  long i = (long)blockIdx.x * 256 + threadIdx.x;
  if (i < n) y[i] = a[i] * b[i];
}

// Xp[i] = [cmf[pep]*smask[prot] (80), feat[prot] (80)]; also writes mp.
__global__ void build_pair_kernel(const int* __restrict__ pep, const int* __restrict__ prot,
                                  const float* __restrict__ cmf, const float* __restrict__ smask,
                                  const float* __restrict__ feat, float* __restrict__ X,
                                  float* __restrict__ mp, int n) {
  long t = (long)blockIdx.x * 256 + threadIdx.x;
  if (t >= (long)n * 160) return;
  int i = (int)(t / 160), d = (int)(t % 160);
  int p = pep[i], q = prot[i];
  float v;
  if (d < 80) {
    v = cmf[(size_t)p * 80 + d] * smask[(size_t)q * 80 + d];
    mp[(size_t)i * 80 + d] = v;
  } else {
    v = feat[(size_t)q * 80 + (d - 80)];
  }
  X[t] = v;
}

// ============================================================================
extern "C" void kernel_launch(void* const* d_in, const int* in_sizes, int n_in,
                              void* d_out, int out_size, void* d_ws, size_t ws_size,
                              hipStream_t stream) {
  (void)in_sizes; (void)n_in; (void)out_size; (void)ws_size;
  auto F = [&](int i) -> const float* { return (const float*)d_in[i]; };
  auto I = [&](int i) -> const int*   { return (const int*)d_in[i]; };

  const int*   atom_cat  = I(0);
  const float* atom_dist = F(1);
  const float* edge_attr = F(2);
  const int*   e_src     = I(3);
  const int*   e_dst     = I(3) + NEDGES;
  const int*   neg_pep   = I(4);
  const int*   neg_prot  = I(5);
  const float* emb       = F(58);
  const int*   pe_src    = I(159);
  const int*   pe_dst    = I(159) + EPEP;
  const float* pep_pre   = F(160);
  const float* posv      = F(161);
  const int*   pos_pep   = I(162);
  const int*   pos_prot  = I(163);
  const float* x_geom    = F(164);
  float* out = (float*)d_out;

  // ---- workspace carve-up: packed weights, then f32 arena (reused) ----
  unsigned short* packBase = (unsigned short*)d_ws;
  const size_t PACK_CAP = 6000000;  // ushorts; actual need ~3.94M (tile-padded)
  size_t packOff = 0;
  auto packW = [&](int wIdx, int K, int N) -> const unsigned short* {
    int nKT = (K + 31) / 32;
    int nCT = (((N + 15) / 16 + 3) / 4) * 4;   // pad col tiles to multiple of 4
    size_t total = (size_t)nCT * nKT * 512;
    unsigned short* dstp = packBase + packOff;
    packOff += total;
    pack_w_kernel<<<(int)((total + 255) / 256), 256, 0, stream>>>(F(wIdx), dstp, K, N, nKT, (long)total);
    return dstp;
  };

  float* fbase = (float*)(packBase + PACK_CAP);
  size_t foff = 0;
  auto alloc = [&](size_t n) -> float* { float* p = fbase + foff; foff += n; return p; };
  float* P0       = alloc(25600000);             // big ping-pong (also XA/XE/HC/pep/Xp)
  float* P1       = alloc(25600000);
  float* P2       = alloc(25600000);
  float* feat     = alloc((size_t)NATOMS * 80);
  float* agg80    = alloc((size_t)NATOMS * 80);
  float* aggB     = alloc((size_t)NATOMS * 80);  // conv1 agg, reused as t80
  float* xbuf     = alloc((size_t)NATOMS * 80);  // feat+agg, reused as t80b / smask
  float* pep_h0   = alloc((size_t)MPEP * 512);
  float* pep_agg  = alloc((size_t)MPEP * 512);
  float* pep_feat = alloc((size_t)MPEP * 80);
  float* cmf      = alloc((size_t)MPEP * 80);
  float* cmask    = alloc((size_t)MPEP * 80);
  float* a_sum    = alloc((size_t)NATOMS * 15);  // reused as chemical 15-out
  float* a2      = alloc((size_t)NATOMS * 15);
  float* feat16   = alloc((size_t)NATOMS * 16);
  float* edist    = alloc(NEDGES);

  // ---- pack all weights (deterministic order/offsets every call) ----
  const unsigned short* aa0 = packW(6, 512, 1024);
  const unsigned short* aa1 = packW(8, 1024, 1024);
  const unsigned short* aa2 = packW(10, 1024, 512);
  const unsigned short* aa3 = packW(12, 512, 80);
  const unsigned short* ana0 = packW(14, 33, 128);
  const unsigned short* ana1 = packW(16, 128, 128);
  const unsigned short* ana2 = packW(18, 128, 128);
  const unsigned short* ana3 = packW(20, 128, 15);
  const unsigned short* anb0 = packW(22, 15, 128);
  const unsigned short* anb1 = packW(24, 128, 128);
  const unsigned short* anb2 = packW(26, 128, 128);
  const unsigned short* anb3 = packW(28, 128, 15);
  const unsigned short* ch0 = packW(30, 16, 256);
  const unsigned short* ch1 = packW(32, 256, 256);
  const unsigned short* ch2 = packW(34, 256, 128);
  const unsigned short* ch3 = packW(36, 128, 15);
  const unsigned short* enp[5][2];
  for (int l = 0; l < 5; ++l) {
    enp[l][0] = packW(38 + 4 * l, 82, 80);
    enp[l][1] = packW(40 + 4 * l, 80, 80);
  }
  const unsigned short* fsW = packW(59, 15, 80);
  const unsigned short* gp[5][4];
  for (int l = 0; l < 5; ++l) {
    int G = 61 + 14 * l;
    gp[l][0] = packW(G + 0, 80, 256);
    gp[l][1] = packW(G + 2, 256, 256);
    gp[l][2] = packW(G + 4, 256, 80);
    gp[l][3] = packW(G + 6, 80, 80);
  }
  const unsigned short* inW  = packW(131, 80, 512);
  const unsigned short* outW = packW(133, 512, 80);
  const unsigned short* pgaW = packW(135, 512, 512);
  const unsigned short* pgbW = packW(137, 512, 512);
  const unsigned short* pr1 = packW(141, 160, 128);
  const unsigned short* pr2 = packW(143, 128, 128);
  const unsigned short* pr3 = packW(145, 128, 1);
  const unsigned short* sm0 = packW(151, 80, 128);
  const unsigned short* sm1 = packW(153, 128, 128);
  const unsigned short* sm2 = packW(155, 128, 128);
  const unsigned short* sm3 = packW(157, 128, 80);

  // ---- launch helpers ----
  auto gemm = [&](const float* A, const float* A2p, const unsigned short* Wp, const float* bias,
                  float* D, int M, int K, int N, int act) {
    int nNT = (N + 15) / 16;
    int nG  = (nNT + 3) / 4;          // 64-col groups, one per wave
    dim3 g((M + 15) / 16, (nG + 3) / 4);
    if (A2p) {
      if (act == ACT_RELU)
        gemm_bf16_wmma_kernel<ACT_RELU, true><<<g, 128, 0, stream>>>(A, A2p, Wp, bias, D, M, K, N);
      else if (act == ACT_SIG)
        gemm_bf16_wmma_kernel<ACT_SIG, true><<<g, 128, 0, stream>>>(A, A2p, Wp, bias, D, M, K, N);
      else
        gemm_bf16_wmma_kernel<ACT_NONE, true><<<g, 128, 0, stream>>>(A, A2p, Wp, bias, D, M, K, N);
    } else {
      if (act == ACT_RELU)
        gemm_bf16_wmma_kernel<ACT_RELU, false><<<g, 128, 0, stream>>>(A, A2p, Wp, bias, D, M, K, N);
      else if (act == ACT_SIG)
        gemm_bf16_wmma_kernel<ACT_SIG, false><<<g, 128, 0, stream>>>(A, A2p, Wp, bias, D, M, K, N);
      else
        gemm_bf16_wmma_kernel<ACT_NONE, false><<<g, 128, 0, stream>>>(A, A2p, Wp, bias, D, M, K, N);
    }
  };
  auto lnorm = [&](const float* X, const float* gam, const float* bet, float* Y, int M, int N, int relu) {
    layernorm_kernel<<<(M + 3) / 4, 128, 0, stream>>>(X, gam, bet, Y, M, N, relu);
  };
  auto fill0 = [&](float* p, long n) {
    fill_kernel<<<(int)((n + 255) / 256), 256, 0, stream>>>(p, 0.0f, n);
  };
  auto segsum = [&](const float* X, const int* s, const int* d, long E, int Dm, float* Agg) {
    long tt = E * (long)Dm;
    seg_gather_scatter_kernel<<<(int)((tt + 255) / 256), 256, 0, stream>>>(X, s, d, E, Dm, Agg);
  };

  // ==================== peptide branch ====================
  gemm(pep_pre, nullptr, aa0, F(7),  P0, MPEP, 512, 1024, ACT_RELU);
  gemm(P0, nullptr, aa1, F(9),  P1, MPEP, 1024, 1024, ACT_RELU);
  gemm(P1, nullptr, aa2, F(11), P0, MPEP, 1024, 512, ACT_RELU);
  gemm(P0, nullptr, aa3, F(13), pep_feat, MPEP, 512, 80, ACT_NONE);
  gemm(pep_feat, nullptr, inW, F(132), pep_h0, MPEP, 80, 512, ACT_NONE);
  fill0(pep_agg, (long)MPEP * 512);
  segsum(pep_h0, pe_src, pe_dst, EPEP, 512, pep_agg);
  gemm(pep_h0, pep_agg, pgaW, F(136), P2, MPEP, 512, 512, ACT_NONE);   // lin_a(h + agg)
  lnorm(P2, F(140), F(139), P0, MPEP, 512, 1);                         // LN + ReLU
  gemm(P0, nullptr, pgbW, F(138), P1, MPEP, 512, 512, ACT_NONE);       // lin_b
  gemm(P1, nullptr, outW, F(134), cmask, MPEP, 512, 80, ACT_NONE);     // context mask
  ew_mul_kernel<<<(int)(((long)MPEP * 80 + 255) / 256), 256, 0, stream>>>(
      pep_feat, cmask, cmf, (long)MPEP * 80);

  // ==================== protein: atom features ====================
  const int RTOT = NATOMS * KNBR, CH_A = 160000;  // 10 chunks
  for (int r0 = 0; r0 < RTOT; r0 += CH_A) {
    build_atom_input_kernel<<<(int)(((long)CH_A * 33 + 255) / 256), 256, 0, stream>>>(
        atom_cat, atom_dist, emb, P0, r0, CH_A);
    gemm(P0, nullptr, ana0, F(15), P1, CH_A, 33, 128, ACT_RELU);
    gemm(P1, nullptr, ana1, F(17), P2, CH_A, 128, 128, ACT_RELU);
    gemm(P2, nullptr, ana2, F(19), P1, CH_A, 128, 128, ACT_RELU);
    gemm(P1, nullptr, ana3, F(21), P0, CH_A, 128, 15, ACT_NONE);
    reduce16_kernel<<<(int)(((long)(CH_A / 16) * 15 + 255) / 256), 256, 0, stream>>>(
        P0, a_sum, r0, CH_A);
  }
  gemm(a_sum, nullptr, anb0, F(23), P0, NATOMS, 15, 128, ACT_RELU);
  gemm(P0, nullptr, anb1, F(25), P1, NATOMS, 128, 128, ACT_RELU);
  gemm(P1, nullptr, anb2, F(27), P0, NATOMS, 128, 128, ACT_RELU);
  gemm(P0, nullptr, anb3, F(29), a2, NATOMS, 128, 15, ACT_NONE);
  build_feat16_kernel<<<(int)(((long)NATOMS * 16 + 255) / 256), 256, 0, stream>>>(
      a2, x_geom, feat16, NATOMS);
  gemm(feat16, nullptr, ch0, F(31), P0, NATOMS, 16, 256, ACT_RELU);
  gemm(P0, nullptr, ch1, F(33), P1, NATOMS, 256, 256, ACT_RELU);
  gemm(P1, nullptr, ch2, F(35), P0, NATOMS, 256, 128, ACT_RELU);
  gemm(P0, nullptr, ch3, F(37), a_sum, NATOMS, 128, 15, ACT_NONE);
  gemm(a_sum, nullptr, fsW, F(60), feat, NATOMS, 15, 80, ACT_NONE);

  edge_dist_kernel<<<(NEDGES + 255) / 256, 256, 0, stream>>>(posv, e_src, e_dst, edist, NEDGES);

  // ==================== protein: 5 GIN layers ====================
  const int CH_E = 200000;  // 8 chunks
  for (int l = 0; l < 5; ++l) {
    int G = 61 + 14 * l, EN = 38 + 4 * l;
    // edge MLP -> segment_sum by src
    fill0(agg80, (long)NATOMS * 80);
    for (int e0 = 0; e0 < NEDGES; e0 += CH_E) {
      build_edge_input_kernel<<<(int)(((long)CH_E * 82 + 255) / 256), 256, 0, stream>>>(
          edist, edge_attr, e_dst, feat, P0, e0, CH_E);
      gemm(P0, nullptr, enp[l][0], F(EN + 1), P1, CH_E, 82, 80, ACT_RELU);
      gemm(P1, nullptr, enp[l][1], F(EN + 3), P2, CH_E, 80, 80, ACT_NONE);
      scatter_rows_kernel<<<(int)(((long)CH_E * 80 + 255) / 256), 256, 0, stream>>>(
          P2, e_src + e0, CH_E, 80, agg80);
    }
    // x = feat + agg; conv1
    ew_add_kernel<<<(int)(((long)NATOMS * 80 + 255) / 256), 256, 0, stream>>>(
        feat, agg80, xbuf, (long)NATOMS * 80);
    fill0(aggB, (long)NATOMS * 80);
    segsum(xbuf, e_src, e_dst, NEDGES, 80, aggB);
    gemm(xbuf, aggB, gp[l][0], F(G + 1), P0, NATOMS, 80, 256, ACT_NONE);   // lin1a(x+agg1)
    lnorm(P0, F(G + 9), F(G + 8), P1, NATOMS, 256, 1);                     // ln1 + relu
    gemm(P1, nullptr, gp[l][1], F(G + 3), P0, NATOMS, 256, 256, ACT_NONE); // lin1b
    lnorm(P0, F(G + 13), F(G + 12), P1, NATOMS, 256, 1);                   // ln_mid + relu
    // conv2
    fill0(P2, (long)NATOMS * 256);
    segsum(P1, e_src, e_dst, NEDGES, 256, P2);
    gemm(P1, P2, gp[l][2], F(G + 5), aggB, NATOMS, 256, 80, ACT_NONE);     // lin2a(h+agg2) -> t80
    lnorm(aggB, F(G + 11), F(G + 10), xbuf, NATOMS, 80, 1);                // ln2 + relu
    gemm(xbuf, nullptr, gp[l][3], F(G + 7), feat, NATOMS, 80, 80, ACT_NONE); // lin2b -> new feat
  }

  // surface mask
  gemm(feat, nullptr, sm0, F(152), P0, NATOMS, 80, 128, ACT_RELU);
  gemm(P0, nullptr, sm1, F(154), P1, NATOMS, 128, 128, ACT_RELU);
  gemm(P1, nullptr, sm2, F(156), P0, NATOMS, 128, 128, ACT_RELU);
  gemm(P0, nullptr, sm3, F(158), xbuf, NATOMS, 128, 80, ACT_NONE);  // smask in xbuf

  // ==================== pairing head ====================
  auto pro_head = [&](const int* pidx, const int* qidx, float* mp_out, float* pred_out) {
    build_pair_kernel<<<(int)(((long)NPAIRS * 160 + 255) / 256), 256, 0, stream>>>(
        pidx, qidx, cmf, xbuf, feat, P0, mp_out, NPAIRS);
    gemm(P0, nullptr, pr1, F(142), P1, NPAIRS, 160, 128, ACT_RELU);
    lnorm(P1, F(148), F(147), P2, NPAIRS, 128, 0);
    gemm(P2, nullptr, pr2, F(144), P1, NPAIRS, 128, 128, ACT_RELU);
    lnorm(P1, F(150), F(149), P2, NPAIRS, 128, 0);
    gemm(P2, nullptr, pr3, F(146), pred_out, NPAIRS, 128, 1, ACT_SIG);
  };
  pro_head(pos_pep, pos_prot, out + 100000, out);
  pro_head(neg_pep, neg_prot, out + 100000 + (long)NPAIRS * 80, out + NPAIRS);
}
